// BidLatte_77283641524846
// MI455X (gfx1250) — matrix operand order repaired and checked by reading, not compiled, not run on hardware
//
#include <hip/hip_runtime.h>
#include <hip/hip_bf16.h>

// ---------------------------------------------------------------------------
// Problem constants (fixed by setup_inputs)
// ---------------------------------------------------------------------------
#define BB 4
#define TT 8192
#define DD 1024
#define LL 128
#define HH 16
#define DH 64
#define LH 8
#define MTOT (BB * TT)   // 32768 rows

typedef __attribute__((ext_vector_type(16))) __bf16        v16bf;
typedef __attribute__((ext_vector_type(8)))  __bf16        v8bf;
typedef __attribute__((ext_vector_type(4)))  __bf16        v4bf;
typedef __attribute__((ext_vector_type(8)))  float         v8f;
typedef __attribute__((ext_vector_type(4)))  unsigned int  v4u;

__device__ __forceinline__ __bf16 f2bf(float f) { return (__bf16)f; }

// Padded leading dims (40 elems = 80 B = 20 banks -> conflict-free b128 reads)
#define LDA_P 40
#define LDB_P 40

// K-swizzle for A tiles: swap bits 3<->4 so a lane's v16bf is contiguous.
// Stored row order: [K0..7 | K16..23 | K8..15 | K24..31]
__device__ __forceinline__ int kperm(int k) {
    return (k & ~24) | ((k & 8) << 1) | ((k & 16) >> 1);
}

// ---- CDNA5 async memory->LDS copy (ASYNCcnt-tracked, per-lane addresses) ---
__device__ __forceinline__ void async_copy_b128(unsigned lds_off,
                                                const void* gaddr) {
    asm volatile("global_load_async_to_lds_b128 %0, %1, off"
                 :: "v"(lds_off),
                    "v"((unsigned long long)(__SIZE_TYPE__)gaddr)
                 : "memory");
}
__device__ __forceinline__ void wait_async0() {
    asm volatile("s_wait_asynccnt 0x0" ::: "memory");
}
// Low 32 bits of a generic pointer to LDS == DS-space byte address (ISA 10.2)
__device__ __forceinline__ unsigned lds_off32(const void* p) {
    return (unsigned)(__SIZE_TYPE__)p;
}

// A fragment (16x32 bf16): lane<16 -> row wrow+lane, bytes 0..31 of row;
// lane>=16 -> row wrow+lane-16, bytes 32..63.  Two ds_load_b128 per lane.
__device__ __forceinline__ v16bf load_a_frag(const __bf16* sA, int wrow) {
    int lane = threadIdx.x & 31;
    const __bf16* p = sA + (wrow + (lane & 15)) * LDA_P + ((lane >> 4) << 4);
    v16bf a;
    ((v4u*)&a)[0] = *(const v4u*)p;
    ((v4u*)&a)[1] = *(const v4u*)(p + 8);
    return a;
}

// B fragment (32x16 bf16, column-major LDS): lane<16 -> col col0+lane K0..7,K16..23;
// lane>=16 -> col col0+lane-16, K8..15,K24..31.  Two ds_load_b128 per lane.
__device__ __forceinline__ v16bf load_b_frag(const __bf16* sB, int col0) {
    int lane = threadIdx.x & 31;
    const __bf16* p = sB + (col0 + (lane & 15)) * LDB_P + ((lane >> 4) << 3);
    v16bf b;
    ((v4u*)&b)[0] = *(const v4u*)p;
    ((v4u*)&b)[1] = *(const v4u*)(p + 16);
    return b;
}

// ---------------------------------------------------------------------------
// Kernel 0: init reduction buffers
// ---------------------------------------------------------------------------
__global__ __launch_bounds__(256) void lat_init(float* kv, float* ksum,
                                                unsigned int* kmax) {
    int i = blockIdx.x * 256 + threadIdx.x;
    if (i < BB * HH * LH * DH) kv[i] = 0.0f;
    if (i < BB * LL) {
        ksum[i] = 0.0f;
        kmax[i] = ~__float_as_uint(-3.0e38f);   // monotonic-uint of -3e38
    }
}

// ---------------------------------------------------------------------------
// Kernel 1: fused QKV projection GEMM (WMMA bf16, f32 acc), double-buffered
//   blockIdx.x : 64-row tile of (B*T)  [512];  blockIdx.y : 0=K,1=Q,2..9=V
// ---------------------------------------------------------------------------
__global__ __launch_bounds__(256) void lat_proj(
    const float* __restrict__ X, const float* __restrict__ mask,
    const float* __restrict__ Wk, const float* __restrict__ Wq,
    const float* __restrict__ Wv,
    float* __restrict__ kscore, __bf16* __restrict__ qsb,
    __bf16* __restrict__ vbuf, unsigned int* __restrict__ kmax) {
    constexpr int BM = 64, BN = 128, BK = 32;
    __shared__ __bf16 sA[2][BM * LDA_P];  // 2 x 5 KB  (K-swizzled rows)
    __shared__ __bf16 sB[2][BN * LDB_P];  // 2 x 10 KB (column-major)
    __shared__ float  sC[BM * BN];        // 32 KB (epilogue tile)
    __shared__ float  sMask[BM];

    const int tid   = threadIdx.x;
    const int chunk = blockIdx.y;
    const int r0    = blockIdx.x * BM;
    const int b     = r0 / TT;
    const int t0    = r0 % TT;

    const float* W;
    int wN, n0;
    if (chunk == 0)      { W = Wk; wN = LL; n0 = 0; }
    else if (chunk == 1) { W = Wq; wN = LL; n0 = 0; }
    else                 { W = Wv; wN = DD; n0 = (chunk - 2) * BN; }

    if (tid < BM) sMask[tid] = mask[(size_t)b * TT + t0 + tid];

    const int wave = tid >> 5;
    const int wrow = (wave & 3) * 16;     // 0,16,32,48
    const int wcol = (wave >> 2) * 64;    // 0,64
    v8f acc[4] = {};

    // A: thread owns row ar, 8 consecutive K   (64 rows x 4 k8-groups = 256)
    const int ar  = tid >> 2;
    const int ak8 = (tid & 3) * 8;
    // B: thread owns 4x4 sub-block (k4, n4)    (8 k-groups x 32 n-groups)
    const int bk4 = (tid & 7) * 4;
    const int bn4 = (tid >> 3) * 4;

    float4 areg[2], breg[4];

    auto stage_load = [&](int k0) {
        areg[0] = *(const float4*)&X[(size_t)(r0 + ar) * DD + k0 + ak8];
        areg[1] = *(const float4*)&X[(size_t)(r0 + ar) * DD + k0 + ak8 + 4];
#pragma unroll
        for (int i = 0; i < 4; ++i)
            breg[i] = *(const float4*)&W[(size_t)(k0 + bk4 + i) * wN + n0 + bn4];
    };
    auto stage_store = [&](int buf) {
        v8bf pa = {f2bf(areg[0].x), f2bf(areg[0].y), f2bf(areg[0].z),
                   f2bf(areg[0].w), f2bf(areg[1].x), f2bf(areg[1].y),
                   f2bf(areg[1].z), f2bf(areg[1].w)};
        *(v8bf*)&sA[buf][ar * LDA_P + kperm(ak8)] = pa;        // one b128
#pragma unroll
        for (int j = 0; j < 4; ++j) {                          // 4x b64
            v4bf pc = {f2bf((&breg[0].x)[j]), f2bf((&breg[1].x)[j]),
                       f2bf((&breg[2].x)[j]), f2bf((&breg[3].x)[j])};
            *(v4bf*)&sB[buf][(bn4 + j) * LDB_P + bk4] = pc;
        }
    };

    stage_load(0);
    stage_store(0);
    __syncthreads();

    int cur = 0;
    for (int k0 = 0; k0 < DD; k0 += BK) {
        const bool more = (k0 + BK) < DD;
        if (more) stage_load(k0 + BK);    // issue global loads early
        v16bf a = load_a_frag(sA[cur], wrow);
#pragma unroll
        for (int j = 0; j < 4; ++j) {
            v16bf bf = load_b_frag(sB[cur], wcol + j * 16);
            acc[j] = __builtin_amdgcn_wmma_f32_16x16x32_bf16(
                false, a, false, bf, (short)0, acc[j], false, false);
        }
        if (more) {
            stage_store(cur ^ 1);         // waits for loads after the WMMAs
            __syncthreads();
        }
        cur ^= 1;
    }

    // spill accumulators to fp32 LDS tile for the epilogues
    {
        int lane = tid & 31;
        int cn = lane & 15, rh = (lane >> 4) * 8;
#pragma unroll
        for (int j = 0; j < 4; ++j)
#pragma unroll
            for (int r = 0; r < 8; ++r)
                sC[(wrow + rh + r) * BN + wcol + j * 16 + cn] = acc[j][r];
    }
    __syncthreads();

    if (chunk == 0) {
        for (int i = tid; i < BM * (LL / 4); i += 256) {
            int r = i / (LL / 4), c4 = (i % (LL / 4)) * 4;
            *(float4*)&kscore[(size_t)(r0 + r) * LL + c4] =
                *(const float4*)&sC[r * BN + c4];
        }
        if (tid < LL) {
            float mx = -3.0e38f;
            for (int r = 0; r < BM; ++r) mx = fmaxf(mx, sC[r * BN + tid]);
            unsigned int u = __float_as_uint(mx);
            u = (u & 0x80000000u) ? ~u : (u | 0x80000000u);
            atomicMax(&kmax[b * LL + tid], u);
        }
    } else if (chunk == 1) {
        for (int g = tid; g < BM * HH; g += 256) {
            int r = g / HH, h = g % HH;
            float v[LH];
            float mx = -3.0e38f;
#pragma unroll
            for (int l = 0; l < LH; ++l) {
                v[l] = sC[r * BN + h * LH + l];
                mx   = fmaxf(mx, v[l]);
            }
            float s = 0.0f;
#pragma unroll
            for (int l = 0; l < LH; ++l) { v[l] = __expf(v[l] - mx); s += v[l]; }
            float inv = 1.0f / s;
            v8bf q;
#pragma unroll
            for (int l = 0; l < LH; ++l) q[l] = f2bf(v[l] * inv);
            *(v8bf*)&qsb[(size_t)(r0 + r) * LL + h * LH] = q;   // 16-B store
        }
    } else {
        for (int i = tid; i < BM * (BN / 4); i += 256) {
            int r = i >> 5, c4 = (i & 31) * 4;
            float4 v = *(const float4*)&sC[r * BN + c4];
            float  m = sMask[r];
            v4bf p = {f2bf(v.x * m), f2bf(v.y * m), f2bf(v.z * m), f2bf(v.w * m)};
            *(v4bf*)&vbuf[(size_t)(r0 + r) * DD + n0 + c4] = p;
        }
    }
}

// ---------------------------------------------------------------------------
// Kernel 2: Kv / Ksum accumulation.  Kv[b,h,l,d] += e[t,h,l] * V[t,h,d]
// ---------------------------------------------------------------------------
__global__ __launch_bounds__(256) void lat_kv(
    const float* __restrict__ kscore, const __bf16* __restrict__ vbuf,
    const float* __restrict__ mask, const unsigned int* __restrict__ kmaxu,
    float* __restrict__ kv, float* __restrict__ ksum) {
    constexpr int CH = 128;
    const int b   = blockIdx.y;
    const int t0  = blockIdx.x * CH;
    const int tid = threadIdx.x;
    __shared__ float sE[LL];
    __shared__ float sV[DD];
    __shared__ float sMax[LL];

    if (tid < LL) {
        unsigned int u = kmaxu[b * LL + tid];
        u = (u & 0x80000000u) ? (u & 0x7FFFFFFFu) : ~u;
        sMax[tid] = __uint_as_float(u);
    }
    float acc[32];
#pragma unroll
    for (int k = 0; k < 32; ++k) acc[k] = 0.0f;
    float ksum_loc = 0.0f;
    __syncthreads();

    for (int t = t0; t < t0 + CH; ++t) {
        const size_t row = (size_t)b * TT + t;
        if (tid < DD / 8) {                       // 128 threads x 8 bf16 (b128)
            v8bf v = *(const v8bf*)&vbuf[row * DD + tid * 8];
#pragma unroll
            for (int j = 0; j < 8; ++j) sV[tid * 8 + j] = (float)v[j];
        }
        if (tid < LL) {
            float e = __expf(kscore[row * LL + tid] - sMax[tid]) * mask[row];
            sE[tid] = e;
            ksum_loc += e;
        }
        __syncthreads();
#pragma unroll
        for (int k = 0; k < 32; ++k) {
            int o    = tid + k * 256;             // o = h*512 + l*64 + d
            int lidx = o >> 6;
            int h    = o >> 9;
            int d    = o & 63;
            acc[k] += sE[lidx] * sV[h * DH + d];
        }
        __syncthreads();
    }
#pragma unroll
    for (int k = 0; k < 32; ++k)
        atomicAdd(&kv[(size_t)b * (HH * LH * DH) + tid + k * 256], acc[k]);
    if (tid < LL) atomicAdd(&ksum[b * LL + tid], ksum_loc);
}

// ---------------------------------------------------------------------------
// Kernel 3: y[t, h*64+d] = sum_l Qs[t,h*8+l] * (Kv/Ksum)[h,l,d]  (bf16 out)
// ---------------------------------------------------------------------------
__global__ __launch_bounds__(256) void lat_y(
    const float* __restrict__ kv, const float* __restrict__ ksum,
    const __bf16* __restrict__ qsb, __bf16* __restrict__ ybuf) {
    const int b   = blockIdx.y;
    const int t0  = blockIdx.x * 128;
    const int tid = threadIdx.x;
    __shared__ float sKv[HH * LH * DH];   // 32 KB
    __shared__ float sQ[LL];

    for (int i = tid; i < HH * LH * DH; i += 256) {
        int lidx = i >> 6;
        sKv[i] = kv[(size_t)b * (HH * LH * DH) + i] / ksum[b * LL + lidx];
    }
    __syncthreads();

    for (int t = t0; t < t0 + 128; ++t) {
        const size_t row = (size_t)b * TT + t;
        if (tid < LL / 8) {                       // 16 threads x 8 bf16 (b128)
            v8bf q = *(const v8bf*)&qsb[row * LL + tid * 8];
#pragma unroll
            for (int j = 0; j < 8; ++j) sQ[tid * 8 + j] = (float)q[j];
        }
        __syncthreads();
        v4bf o;
#pragma unroll
        for (int j = 0; j < 4; ++j) {
            int dg = tid * 4 + j;
            int h  = dg >> 6, d = dg & 63;
            float s = 0.0f;
#pragma unroll
            for (int l = 0; l < LH; ++l)
                s += sQ[h * LH + l] * sKv[(h * LH + l) * DH + d];
            o[j] = f2bf(s);
        }
        *(v4bf*)&ybuf[row * DD + tid * 4] = o;    // 8-B store
        __syncthreads();
    }
}

// ---------------------------------------------------------------------------
// Kernel 4: out = y @ o_proj (WMMA bf16 -> fp32), double-buffered,
//           A tile streamed with GLOBAL_LOAD_ASYNC_TO_LDS_B128 (ASYNCcnt)
// ---------------------------------------------------------------------------
__global__ __launch_bounds__(256) void lat_out(
    const __bf16* __restrict__ ybuf, const float* __restrict__ o_proj,
    float* __restrict__ out) {
    constexpr int BM = 128, BN = 128, BK = 32;
    __shared__ __bf16 sA[2][BM * LDA_P];  // 2 x 10 KB
    __shared__ __bf16 sB[2][BN * LDB_P];  // 2 x 10 KB
    const int tid = threadIdx.x;
    const int r0  = blockIdx.x * BM;
    const int n0  = blockIdx.y * BN;
    const int wave = tid >> 5;
    const int wrow = (wave & 3) * 32;     // 0,32,64,96
    const int wcol = (wave >> 2) * 64;    // 0,64
    v8f acc[2][4] = {};

    // B: thread owns 4x4 sub-block (k4, n4)
    const int bk4 = (tid & 7) * 4;
    const int bn4 = (tid >> 3) * 4;
    float4 breg[4];

    // A: pure address-swizzled b128 copies -> async direct to LDS
    auto stage_A_async = [&](int k0, int buf) {
#pragma unroll
        for (int u = 0; u < 2; ++u) {     // 128*4 b128 groups / 256 threads
            int g = tid + u * 256;
            int r = g >> 2, k8 = (g & 3) * 8;
            async_copy_b128(lds_off32(&sA[buf][r * LDA_P + kperm(k8)]),
                            &ybuf[(size_t)(r0 + r) * DD + k0 + k8]);
        }
    };
    auto stage_B_load = [&](int k0) {
#pragma unroll
        for (int i = 0; i < 4; ++i)
            breg[i] = *(const float4*)&o_proj[(size_t)(k0 + bk4 + i) * DD +
                                              n0 + bn4];
    };
    auto stage_B_store = [&](int buf) {
#pragma unroll
        for (int j = 0; j < 4; ++j) {                          // 4x b64
            v4bf pc = {f2bf((&breg[0].x)[j]), f2bf((&breg[1].x)[j]),
                       f2bf((&breg[2].x)[j]), f2bf((&breg[3].x)[j])};
            *(v4bf*)&sB[buf][(bn4 + j) * LDB_P + bk4] = pc;
        }
    };

    stage_A_async(0, 0);
    stage_B_load(0);
    stage_B_store(0);
    wait_async0();
    __syncthreads();

    int cur = 0;
    for (int k0 = 0; k0 < DD; k0 += BK) {
        const bool more = (k0 + BK) < DD;
        if (more) {
            stage_A_async(k0 + BK, cur ^ 1);   // async: no wave stall
            stage_B_load(k0 + BK);             // issue global loads early
        }
        v16bf a0 = load_a_frag(sA[cur], wrow);
        v16bf a1 = load_a_frag(sA[cur], wrow + 16);
#pragma unroll
        for (int j = 0; j < 4; ++j) {
            v16bf bf = load_b_frag(sB[cur], wcol + j * 16);
            acc[0][j] = __builtin_amdgcn_wmma_f32_16x16x32_bf16(
                false, a0, false, bf, (short)0, acc[0][j], false, false);
            acc[1][j] = __builtin_amdgcn_wmma_f32_16x16x32_bf16(
                false, a1, false, bf, (short)0, acc[1][j], false, false);
        }
        if (more) {
            stage_B_store(cur ^ 1);            // load-wait lands after WMMAs
            wait_async0();                     // A-tile async writes visible
            __syncthreads();
        }
        cur ^= 1;
    }

    int lane = tid & 31;
    int cn = lane & 15, rh = (lane >> 4) * 8;
#pragma unroll
    for (int i = 0; i < 2; ++i)
#pragma unroll
        for (int j = 0; j < 4; ++j)
#pragma unroll
            for (int r = 0; r < 8; ++r)
                out[(size_t)(r0 + wrow + i * 16 + rh + r) * DD +
                    n0 + wcol + j * 16 + cn] = acc[i][j][r];
}

// ---------------------------------------------------------------------------
// Launch
// ---------------------------------------------------------------------------
extern "C" void kernel_launch(void* const* d_in, const int* in_sizes, int n_in,
                              void* d_out, int out_size, void* d_ws,
                              size_t ws_size, hipStream_t stream) {
    const float* X      = (const float*)d_in[0];
    const float* amask  = (const float*)d_in[1];
    const float* Wk     = (const float*)d_in[2];
    const float* Wq     = (const float*)d_in[3];
    const float* Wv     = (const float*)d_in[4];
    const float* o_proj = (const float*)d_in[5];
    float* out = (float*)d_out;

    char* p = (char*)d_ws;
    auto take = [&](size_t bytes) {
        char* q = p;
        p += (bytes + 255) & ~(size_t)255;
        return (void*)q;
    };
    float*        kscore = (float*)take((size_t)MTOT * LL * sizeof(float));
    __bf16*       qsb    = (__bf16*)take((size_t)MTOT * LL * 2);
    __bf16*       vbuf   = (__bf16*)take((size_t)MTOT * DD * 2);
    __bf16*       ybuf   = (__bf16*)take((size_t)MTOT * DD * 2);
    float*        kv     = (float*)take((size_t)BB * HH * LH * DH * 4);
    float*        ksum   = (float*)take((size_t)BB * LL * 4);
    unsigned int* kmax   = (unsigned int*)take((size_t)BB * LL * 4);

    lat_init<<<128, 256, 0, stream>>>(kv, ksum, kmax);
    lat_proj<<<dim3(MTOT / 64, 10), 256, 0, stream>>>(X, amask, Wk, Wq, Wv,
                                                      kscore, qsb, vbuf, kmax);
    lat_kv<<<dim3(TT / 128, BB), 256, 0, stream>>>(kscore, vbuf, amask, kmax,
                                                   kv, ksum);
    lat_y<<<dim3(TT / 128, BB), 256, 0, stream>>>(kv, ksum, qsb, ybuf);
    lat_out<<<dim3(MTOT / 128, DD / 128), 256, 0, stream>>>(ybuf, o_proj, out);
}
